// AttentionSecondOrderLSTMCell_64398739636286
// MI455X (gfx1250) — matrix operand order repaired and checked by reading, not compiled, not run on hardware
//
#include <hip/hip_runtime.h>

// ---------------------------------------------------------------------------
// AttentionSecondOrderLSTMCell fused kernel for gfx1250 (MI455X, wave32, WMMA)
// ---------------------------------------------------------------------------
// d_in order: x(16384x128) temperature(1) h0(16384x128) c0(16384x128)
//             W_ih(16x512x128) W_hh(16x512x128) b_ih(16x512) b_hh(16x512)
//             V(16x128)
// d_out: updated_hidden (16384x128 f32) ++ updated_cell (16384x128 f32)
// d_ws : 4 MB bf16 weight pack in WMMA B-fragment order
// ---------------------------------------------------------------------------

typedef __attribute__((ext_vector_type(16))) __bf16 v16bf;
typedef __attribute__((ext_vector_type(8)))  float  v8f;

struct alignas(16) U4 { unsigned int x, y, z, w; };

union Frag16 {
  v16bf v;
  unsigned int u[8];
};

__device__ __forceinline__ unsigned short f32_to_bf16(float f) {
  unsigned int u = __float_as_uint(f);
  u += 0x7FFFu + ((u >> 16) & 1u);   // round-to-nearest-even
  return (unsigned short)(u >> 16);
}

__device__ __forceinline__ unsigned int pack_bf16x2(float lo, float hi) {
  return (unsigned int)f32_to_bf16(lo) | ((unsigned int)f32_to_bf16(hi) << 16);
}

__device__ __forceinline__ float sigmoid_f(float z) {
  return __builtin_amdgcn_rcpf(1.0f + __expf(-z));
}
__device__ __forceinline__ float tanh_f(float z) {
  // tanh(z) = 1 - 2/(1 + e^{2z})
  return 1.0f - 2.0f * __builtin_amdgcn_rcpf(1.0f + __expf(2.0f * z));
}

__device__ __forceinline__ void load_frag(Frag16& f, const unsigned short* p) {
  const U4* q = reinterpret_cast<const U4*>(p);
  const U4 q0 = q[0];
  const U4 q1 = q[1];
  f.u[0] = q0.x; f.u[1] = q0.y; f.u[2] = q0.z; f.u[3] = q0.w;
  f.u[4] = q1.x; f.u[5] = q1.y; f.u[6] = q1.z; f.u[7] = q1.w;
}

// ---------------------------------------------------------------------------
// Kernel 1: pack W_ih / W_hh (f32) into bf16 WMMA B-fragments.
// Layout (ushort units):
//   off = ht*262144 + k*16384 + gate*4096 + src*2048 + kb*512 + lane*16 + e
// where fragment element e of lane L maps to
//   row g = gate*128 + ht*16 + (L%16)  (B column N)
//   i     = kb*32 + 8*(L/16) + e + (e&8)  (contraction K)
// ---------------------------------------------------------------------------
__global__ __launch_bounds__(256) void pack_weights_kernel(
    const float* __restrict__ W_ih, const float* __restrict__ W_hh,
    unsigned short* __restrict__ out) {
  const unsigned int t = blockIdx.x * 256u + threadIdx.x;  // 0 .. 2M-1
  const unsigned int e    =  t        & 15u;
  const unsigned int lane = (t >> 4)  & 31u;
  const unsigned int kb   = (t >> 9)  & 3u;
  const unsigned int src  = (t >> 11) & 1u;
  const unsigned int gate = (t >> 12) & 3u;
  const unsigned int k    = (t >> 14) & 15u;
  const unsigned int ht   =  t >> 18;
  const float* W = src ? W_hh : W_ih;
  const unsigned int g = gate * 128u + ht * 16u + (lane & 15u);
  const unsigned int i = kb * 32u + ((lane >> 4) << 3) + e + (e & 8u);
  out[t] = f32_to_bf16(W[(k * 512u + g) * 128u + i]);
}

// ---------------------------------------------------------------------------
// Kernel 2: fused attention + 16 LSTM cells + weighted combine.
// grid = B/16 blocks, block = 256 threads (8 waves).
// Wave `ht` owns hidden columns [ht*16, ht*16+16) and loops over all cells.
// ---------------------------------------------------------------------------
__global__ __launch_bounds__(256) void lstm_fused_kernel(
    const float* __restrict__ x, const float* __restrict__ temperature,
    const float* __restrict__ h0, const float* __restrict__ c0,
    const unsigned short* __restrict__ packedW,
    const float* __restrict__ b_ih, const float* __restrict__ b_hh,
    const float* __restrict__ V,
    float* __restrict__ outH, float* __restrict__ outC) {
  __shared__ float s_scores[16][16];
  __shared__ float s_alpha[16][16];

  const int tid  = threadIdx.x;
  const int lane = tid & 31;
  const int wave = tid >> 5;       // ht: hidden column tile 0..7
  const int half = lane >> 4;
  const int m16  = lane & 15;
  const int row_base = blockIdx.x << 4;
  const int col = wave * 16 + m16; // hidden/gate column owned by this lane

  // ---- 1) attention scores: score[m][k] = x[m] . V[k] -------------------
  {
    const int sm = tid & 15;
    const int sk = tid >> 4;
    const float4* xv = reinterpret_cast<const float4*>(x + (size_t)(row_base + sm) * 128);
    const float4* vv = reinterpret_cast<const float4*>(V + sk * 128);
    float s = 0.0f;
#pragma unroll
    for (int j = 0; j < 32; ++j) {
      const float4 a = xv[j];
      const float4 b = vv[j];
      s += a.x * b.x + a.y * b.y + a.z * b.z + a.w * b.w;
    }
    s_scores[sm][sk] = s;
  }
  __syncthreads();

  // ---- 2) softmax over cells (per batch row) -----------------------------
  if (tid < 16) {
    const float invT = __builtin_amdgcn_rcpf(temperature[0]);
    float z[16];
    float mx = -3.0e38f;
#pragma unroll
    for (int kk = 0; kk < 16; ++kk) {
      z[kk] = s_scores[tid][kk] * invT;
      mx = fmaxf(mx, z[kk]);
    }
    float sum = 0.0f;
#pragma unroll
    for (int kk = 0; kk < 16; ++kk) {
      const float e = __expf(z[kk] - mx);
      z[kk] = e;
      sum += e;
    }
    const float inv = __builtin_amdgcn_rcpf(sum);
#pragma unroll
    for (int kk = 0; kk < 16; ++kk) s_alpha[tid][kk] = z[kk] * inv;
  }
  __syncthreads();

  // ---- 3) build bf16 A-fragments for x and h0 (16x32 per kblk) -----------
  // A layout: lane L holds row M=L%16; VGPR j covers K = (j<4 ? 2j : 16+2(j-4)) + 8*(L/16)
  Frag16 ax[4], ah[4];
  {
    const float* xr = x  + (size_t)(row_base + m16) * 128;
    const float* hr = h0 + (size_t)(row_base + m16) * 128;
#pragma unroll
    for (int kb = 0; kb < 4; ++kb) {
      const int i0 = kb * 32 + half * 8;
#pragma unroll
      for (int j = 0; j < 4; ++j) {
        ax[kb].u[j]     = pack_bf16x2(xr[i0 + 2 * j],      xr[i0 + 2 * j + 1]);
        ax[kb].u[j + 4] = pack_bf16x2(xr[i0 + 16 + 2 * j], xr[i0 + 16 + 2 * j + 1]);
        ah[kb].u[j]     = pack_bf16x2(hr[i0 + 2 * j],      hr[i0 + 2 * j + 1]);
        ah[kb].u[j + 4] = pack_bf16x2(hr[i0 + 16 + 2 * j], hr[i0 + 16 + 2 * j + 1]);
      }
    }
  }

  // ---- 4) c0 tile in C/D accumulator layout ------------------------------
  float c0t[8];
#pragma unroll
  for (int r = 0; r < 8; ++r)
    c0t[r] = c0[(size_t)(row_base + r + 8 * half) * 128 + col];

  v8f accH = {0.f, 0.f, 0.f, 0.f, 0.f, 0.f, 0.f, 0.f};
  v8f accC = {0.f, 0.f, 0.f, 0.f, 0.f, 0.f, 0.f, 0.f};

  const unsigned short* wht = packedW + (size_t)wave * 262144u;

  // ---- 5) loop over the 16 second-order cells ----------------------------
#pragma unroll 2
  for (int k = 0; k < 16; ++k) {
    const unsigned short* wk = wht + (size_t)k * 16384u;
    if (k < 15)  // pull next cell's weight block toward L2/L0
      __builtin_prefetch(wk + 16384 + lane * 512, 0, 1);

    // per-row attention weights + per-gate combined biases (issued early)
    float al[8];
#pragma unroll
    for (int r = 0; r < 8; ++r) al[r] = s_alpha[r + 8 * half][k];
    float bias[4];
#pragma unroll
    for (int gate = 0; gate < 4; ++gate)
      bias[gate] = b_ih[k * 512 + gate * 128 + col] +
                   b_hh[k * 512 + gate * 128 + col];

    v8f g[4];
#pragma unroll
    for (int gate = 0; gate < 4; ++gate) {
      const unsigned short* wg = wk + gate * 4096;

      // Stage the whole gate's B fragments first (16 global_load_b128 in
      // flight) so the dependent 8-deep WMMA chain overlaps the next gate's
      // loads instead of waiting loadcnt==0 per fragment.
      Frag16 bx[4], bh[4];
#pragma unroll
      for (int kb = 0; kb < 4; ++kb) {
        load_frag(bx[kb], wg + kb * 512 + lane * 16);
        load_frag(bh[kb], wg + 2048 + kb * 512 + lane * 16);
      }

      v8f acc = {};  // zero C folds to inline constant 0 in the WMMA encoding
#pragma unroll
      for (int kb = 0; kb < 4; ++kb)
        acc = __builtin_amdgcn_wmma_f32_16x16x32_bf16(
            false, ax[kb].v, false, bx[kb].v, (short)0, acc, false, false);
#pragma unroll
      for (int kb = 0; kb < 4; ++kb)
        acc = __builtin_amdgcn_wmma_f32_16x16x32_bf16(
            false, ah[kb].v, false, bh[kb].v, (short)0, acc, false, false);
      g[gate] = acc;
    }

    // gate nonlinearities + cell update + alpha-weighted accumulation
    // (bias folded in here; TRANS ops co-execute with the WMMA pipe)
#pragma unroll
    for (int r = 0; r < 8; ++r) {
      const float ig = sigmoid_f(g[0][r] + bias[0]);
      const float fg = sigmoid_f(g[1][r] + bias[1]);
      const float gg = tanh_f(g[2][r] + bias[2]);
      const float og = sigmoid_f(g[3][r] + bias[3]);
      const float cn = fg * c0t[r] + ig * gg;
      const float hn = og * tanh_f(cn);
      accC[r] += al[r] * cn;
      accH[r] += al[r] * hn;
    }
  }

  // ---- 6) write final h, c ----------------------------------------------
#pragma unroll
  for (int r = 0; r < 8; ++r) {
    const size_t row = (size_t)(row_base + r + 8 * half);
    outH[row * 128 + col] = accH[r];
    outC[row * 128 + col] = accC[r];
  }
}

// ---------------------------------------------------------------------------
extern "C" void kernel_launch(void* const* d_in, const int* in_sizes, int n_in,
                              void* d_out, int out_size, void* d_ws, size_t ws_size,
                              hipStream_t stream) {
  (void)in_sizes; (void)n_in; (void)out_size; (void)ws_size;
  const float* x    = (const float*)d_in[0];
  const float* temp = (const float*)d_in[1];
  const float* h0   = (const float*)d_in[2];
  const float* c0   = (const float*)d_in[3];
  const float* W_ih = (const float*)d_in[4];
  const float* W_hh = (const float*)d_in[5];
  const float* b_ih = (const float*)d_in[6];
  const float* b_hh = (const float*)d_in[7];
  const float* V    = (const float*)d_in[8];

  float* outH = (float*)d_out;
  float* outC = outH + (size_t)16384 * 128;
  unsigned short* packedW = (unsigned short*)d_ws;  // 4 MB bf16 pack

  // 2M packed bf16 elements, one per thread
  pack_weights_kernel<<<2097152 / 256, 256, 0, stream>>>(W_ih, W_hh, packedW);

  // one block per 16 batch rows
  lstm_fused_kernel<<<16384 / 16, 256, 0, stream>>>(
      x, temp, h0, c0, packedW, b_ih, b_hh, V, outH, outC);
}
